// BinaryMultiHeadAttention_37245956391176
// MI455X (gfx1250) — 16-bit and 8-bit matrix operand order repaired and checked by reading, compiled, not run on hardware
//
#include <hip/hip_runtime.h>

typedef __attribute__((ext_vector_type(16))) _Float16 v16h;
typedef __attribute__((ext_vector_type(8)))  _Float16 v8h;
typedef __attribute__((ext_vector_type(8)))  float    v8f;
typedef __attribute__((ext_vector_type(8)))  int      v8i;

namespace cfg {
constexpr int B = 2, S = 2048, DM = 1024, H = 16, HD = 64;
constexpr int M = B * S;
constexpr float QK_SCALE = 0.25f;
}

enum ProjMode { OUT_I8 = 0, OUT_F16T = 1, OUT_F32 = 2 };

union V16HU { v16h v; v8h h[2]; };
union V8IU  { v8i v; uint2 d2[4]; uint4 d4[2]; };

typedef const void __attribute__((address_space(3))) * lds_cvptr;

static __device__ __forceinline__ void async_copy_b128(const void* lds_dst,
                                                       const void* gsrc) {
  const uint32_t loff = (uint32_t)(size_t)(lds_cvptr)lds_dst;
  asm volatile("global_load_async_to_lds_b128 %0, %1, off"
               :: "v"(loff), "v"(gsrc)
               : "memory");
}

__global__ __launch_bounds__(256) void cvt_f32_to_f16_kernel(
    const float* __restrict__ in, _Float16* __restrict__ out, int n) {
  const int i = (blockIdx.x * 256 + threadIdx.x) * 8;
  if (i >= n) return;
  const float4 a = *(const float4*)(in + i);
  const float4 b = *(const float4*)(in + i + 4);
  v8h o;
  o[0] = (_Float16)a.x; o[1] = (_Float16)a.y; o[2] = (_Float16)a.z; o[3] = (_Float16)a.w;
  o[4] = (_Float16)b.x; o[5] = (_Float16)b.y; o[6] = (_Float16)b.z; o[7] = (_Float16)b.w;
  *(v8h*)(out + i) = o;
}

__global__ __launch_bounds__(256) void binary_proj_kernel(
    const _Float16* __restrict__ X,
    const _Float16* __restrict__ Wh,
    const float*    __restrict__ bias,
    void*           __restrict__ outp,
    int mode)
{
  using namespace cfg;
  __shared__ __align__(16) _Float16 Atile[2][64][32];
  __shared__ __align__(16) _Float16 Btile[2][128][32];

  const int tid  = threadIdx.x;
  const int lane = tid & 31;
  const int wave = tid >> 5;
  const int wy   = wave & 3;
  const int wx   = wave >> 2;
  const int ln16 = lane & 15;
  const int hi   = lane >> 4;
  const int m0   = blockIdx.x * 64;
  const int n0   = blockIdx.y * 128;

  v8f acc[4] = {};

  const int arow = tid >> 2, acol = (tid & 3) * 8;
  const int brow = tid >> 1, bcol = (tid & 1) * 16;
  const _Float16* aSrc = X  + (size_t)(m0 + arow) * DM + acol;
  const _Float16* bSrc = Wh + (size_t)(n0 + brow) * DM + bcol;

  auto issue_tile = [&](int k0, int bufi) {
    async_copy_b128(&Atile[bufi][arow][acol],     aSrc + k0);
    async_copy_b128(&Btile[bufi][brow][bcol],     bSrc + k0);
    async_copy_b128(&Btile[bufi][brow][bcol + 8], bSrc + k0 + 8);
  };

  constexpr int T = DM / 32;
  issue_tile(0, 0);

  for (int t = 0; t < T; ++t) {
    const int bufi = t & 1;
    if (t + 1 < T) {
      issue_tile((t + 1) * 32, (t + 1) & 1);
      asm volatile("s_wait_asynccnt 3" ::: "memory");
    } else {
      asm volatile("s_wait_asynccnt 0" ::: "memory");
    }
    __syncthreads();

    V16HU Af;
    const int am = wy * 16 + ln16;
    const int ak = hi ? 8 : 0;
    Af.h[0] = *(const v8h*)&Atile[bufi][am][ak];
    Af.h[1] = *(const v8h*)&Atile[bufi][am][ak + 16];

    V16HU Bf[4];
    #pragma unroll
    for (int nt = 0; nt < 4; ++nt) {
      const int bn = wx * 64 + nt * 16 + ln16;
      const int bk = hi ? 8 : 0;
      Bf[nt].h[0] = *(const v8h*)&Btile[bufi][bn][bk];
      Bf[nt].h[1] = *(const v8h*)&Btile[bufi][bn][bk + 16];
    }
    #pragma unroll
    for (int nt = 0; nt < 4; ++nt)
      acc[nt] = __builtin_amdgcn_wmma_f32_16x16x32_f16(
          false, Af.v, false, Bf[nt].v, (short)0, acc[nt], false, false);

    __syncthreads();
  }

  #pragma unroll
  for (int nt = 0; nt < 4; ++nt) {
    const int n  = n0 + wx * 64 + nt * 16 + ln16;
    const float bn = bias[n];
    #pragma unroll
    for (int r = 0; r < 8; ++r) {
      const int m = m0 + wy * 16 + r + hi * 8;
      const float bit = (acc[nt][r] + bn) > 0.5f ? 1.0f : 0.0f;
      if (mode == OUT_I8) {
        ((unsigned char*)outp)[(size_t)m * DM + n] = (unsigned char)bit;
      } else if (mode == OUT_F16T) {
        const int bb = m >> 11;
        const int s  = m & (S - 1);
        ((_Float16*)outp)[((size_t)(bb * DM + n)) * S + s] = (_Float16)bit;
      } else {
        ((float*)outp)[(size_t)m * DM + n] = bit;
      }
    }
  }
}

__global__ __launch_bounds__(256) void attn_kernel(
    const unsigned char* __restrict__ Qb,
    const unsigned char* __restrict__ Kb,
    const _Float16*      __restrict__ VT,
    _Float16*            __restrict__ out)
{
  using namespace cfg;
  __shared__ __align__(16) _Float16 Plds[8][16][32];

  const int lane = threadIdx.x & 31;
  const int wv   = threadIdx.x >> 5;
  const int gw   = blockIdx.x * 8 + wv;
  const int qt   = gw & (S / 16 - 1);
  const int h    = (gw >> 7) & (H - 1);
  const int b    = gw >> 11;
  const int ln16 = lane & 15;
  const int hi   = lane >> 4;

  V8IU Af;
  {
    const unsigned char* qrow =
        Qb + ((size_t)(b * S + qt * 16 + ln16)) * DM + h * HD + (hi ? 16 : 0);
    #pragma unroll
    for (int c = 0; c < 4; ++c)
      Af.d2[c] = *(const uint2*)(qrow + (c & 1) * 8 + (c >> 1) * 32);
  }

  float mrow[8], lrow[8];
  v8f acc[4] = {};
  #pragma unroll
  for (int r = 0; r < 8; ++r) { mrow[r] = -3.0e38f; lrow[r] = 0.0f; }

  const v8i czero = {0, 0, 0, 0, 0, 0, 0, 0};

  for (int c0 = 0; c0 < S; c0 += 32) {
    v8i sc[2];
    #pragma unroll
    for (int t = 0; t < 2; ++t) {
      const int kt = c0 + t * 16;
      V8IU Bf;
      const unsigned char* krow =
          Kb + ((size_t)(b * S + kt + ln16)) * DM + h * HD + (hi ? 16 : 0);
      Bf.d4[0] = *(const uint4*)(krow);
      Bf.d4[1] = *(const uint4*)(krow + 32);
      sc[t] = __builtin_amdgcn_wmma_i32_16x16x64_iu8(
          false, Af.v, false, Bf.v, czero, false, false);
    }

    float e0[8], e1[8], cmax[8];
    #pragma unroll
    for (int r = 0; r < 8; ++r) {
      e0[r] = (float)sc[0][r] * QK_SCALE;
      e1[r] = (float)sc[1][r] * QK_SCALE;
      cmax[r] = fmaxf(e0[r], e1[r]);
    }
    #pragma unroll
    for (int msk = 1; msk < 16; msk <<= 1) {
      #pragma unroll
      for (int r = 0; r < 8; ++r)
        cmax[r] = fmaxf(cmax[r], __shfl_xor(cmax[r], msk, 32));
    }
    float rs[8];
    #pragma unroll
    for (int r = 0; r < 8; ++r) {
      const float mn    = fmaxf(mrow[r], cmax[r]);
      const float alpha = __expf(mrow[r] - mn);
      mrow[r] = mn;
      lrow[r] *= alpha;
      #pragma unroll
      for (int nt = 0; nt < 4; ++nt) acc[nt][r] *= alpha;
      e0[r] = __expf(e0[r] - mn);
      e1[r] = __expf(e1[r] - mn);
      rs[r] = e0[r] + e1[r];
    }
    #pragma unroll
    for (int msk = 1; msk < 16; msk <<= 1) {
      #pragma unroll
      for (int r = 0; r < 8; ++r)
        rs[r] += __shfl_xor(rs[r], msk, 32);
    }
    #pragma unroll
    for (int r = 0; r < 8; ++r) lrow[r] += rs[r];

    #pragma unroll
    for (int r = 0; r < 8; ++r) {
      const int row = r + hi * 8;
      Plds[wv][row][ln16]      = (_Float16)e0[r];
      Plds[wv][row][16 + ln16] = (_Float16)e1[r];
    }
    asm volatile("s_wait_dscnt 0" ::: "memory");

    V16HU Pf;
    {
      const int ko = hi ? 8 : 0;
      Pf.h[0] = *(const v8h*)&Plds[wv][ln16][ko];
      Pf.h[1] = *(const v8h*)&Plds[wv][ln16][ko + 16];
    }

    #pragma unroll
    for (int nt = 0; nt < 4; ++nt) {
      V16HU Vf;
      const _Float16* vp =
          VT + ((size_t)(b * DM) + h * HD + nt * 16 + ln16) * S + c0 + (hi ? 8 : 0);
      Vf.h[0] = *(const v8h*)vp;
      Vf.h[1] = *(const v8h*)(vp + 16);
      acc[nt] = __builtin_amdgcn_wmma_f32_16x16x32_f16(
          false, Pf.v, false, Vf.v, (short)0, acc[nt], false, false);
    }
  }

  #pragma unroll
  for (int nt = 0; nt < 4; ++nt) {
    #pragma unroll
    for (int r = 0; r < 8; ++r) {
      const int srow = qt * 16 + r + hi * 8;
      const int d    = nt * 16 + ln16;
      out[((size_t)(b * S + srow)) * DM + h * HD + d] =
          (_Float16)(acc[nt][r] / lrow[r]);
    }
  }
}

extern "C" void kernel_launch(void* const* d_in, const int* in_sizes, int n_in,
                              void* d_out, int out_size, void* d_ws, size_t ws_size,
                              hipStream_t stream) {
  using namespace cfg;
  const float* x  = (const float*)d_in[0];
  const float* wq = (const float*)d_in[1];
  const float* bq = (const float*)d_in[2];
  const float* wk = (const float*)d_in[3];
  const float* bk = (const float*)d_in[4];
  const float* wv = (const float*)d_in[5];
  const float* bv = (const float*)d_in[6];
  const float* wo = (const float*)d_in[7];
  const float* bo = (const float*)d_in[8];

  char* ws = (char*)d_ws;
  size_t off = 0;
  _Float16* Xh  = (_Float16*)(ws + off); off += (size_t)M * DM * 2;
  _Float16* Whq = (_Float16*)(ws + off); off += (size_t)DM * DM * 2;
  _Float16* Whk = (_Float16*)(ws + off); off += (size_t)DM * DM * 2;
  _Float16* Whv = (_Float16*)(ws + off); off += (size_t)DM * DM * 2;
  _Float16* Who = (_Float16*)(ws + off); off += (size_t)DM * DM * 2;
  unsigned char* Qi8 = (unsigned char*)(ws + off); off += (size_t)M * DM;
  unsigned char* Ki8 = (unsigned char*)(ws + off); off += (size_t)M * DM;
  _Float16* VT   = (_Float16*)(ws + off); off += (size_t)B * DM * S * 2;
  _Float16* attH = (_Float16*)(ws + off); off += (size_t)M * DM * 2;

  {
    const int nx = M * DM, nw = DM * DM;
    cvt_f32_to_f16_kernel<<<nx / 8 / 256, 256, 0, stream>>>(x,  Xh,  nx);
    cvt_f32_to_f16_kernel<<<nw / 8 / 256, 256, 0, stream>>>(wq, Whq, nw);
    cvt_f32_to_f16_kernel<<<nw / 8 / 256, 256, 0, stream>>>(wk, Whk, nw);
    cvt_f32_to_f16_kernel<<<nw / 8 / 256, 256, 0, stream>>>(wv, Whv, nw);
    cvt_f32_to_f16_kernel<<<nw / 8 / 256, 256, 0, stream>>>(wo, Who, nw);
  }

  dim3 pgrid(M / 64, DM / 128);
  dim3 pblk(256);
  binary_proj_kernel<<<pgrid, pblk, 0, stream>>>(Xh, Whq, bq, (void*)Qi8, OUT_I8);
  binary_proj_kernel<<<pgrid, pblk, 0, stream>>>(Xh, Whk, bk, (void*)Ki8, OUT_I8);
  binary_proj_kernel<<<pgrid, pblk, 0, stream>>>(Xh, Whv, bv, (void*)VT,  OUT_F16T);

  const int totalWaves = B * H * (S / 16);
  attn_kernel<<<totalWaves / 8, 256, 0, stream>>>(Qi8, Ki8, VT, attH);

  binary_proj_kernel<<<pgrid, pblk, 0, stream>>>(attH, Who, bo, d_out, OUT_F32);
}
